// Wav2TensorCore_87917980549676
// MI455X (gfx1250) — matrix operand order repaired and checked by reading, not compiled, not run on hardware
//
#include <hip/hip_runtime.h>
#include <hip/hip_bf16.h>

// ---------------- problem constants (from reference) ----------------
#define BATCH    16
#define T_IN     480000
#define NFFT     1024
#define HOP      256
#define PADH     512          // NFFT/2 reflect pad
#define NFRAMES  1876         // 1 + (T_IN + 2*PADH - NFFT)/HOP
#define FBINS    513          // NFFT/2 + 1
#define NPAD     576          // 9 * 64, padded freq dim for twiddle table
#define CHUNK_ELEMS (NPAD * 32)   // one 32-K chunk of the twiddle table

typedef __attribute__((ext_vector_type(16))) _Float16 v16h;
typedef __attribute__((ext_vector_type(8)))  float    v8f;

union FragAB { float4 f4[2]; v16h v; };

// ---------------- kernel 0: windowed DFT basis (f16), chunk-transposed ------
// Layout: W[(k/32)*NPAD*32 + f*32 + (k%32)] so that the (32K x 64N) tile a
// block needs per K-step is one contiguous 4KB region (async-copy friendly).
// Wr =  hann(n)*cos(2*pi*n*f/NFFT),  Wi = -hann(n)*sin(2*pi*n*f/NFFT)
__global__ __launch_bounds__(256) void init_twiddles(_Float16* __restrict__ Wr,
                                                     _Float16* __restrict__ Wi) {
    int idx = blockIdx.x * 256 + threadIdx.x;
    if (idx >= NFFT * NPAD) return;
    int c   = idx / CHUNK_ELEMS;          // K chunk
    int rem = idx - c * CHUNK_ELEMS;
    int f   = rem >> 5;
    int kl  = rem & 31;
    int n   = (c << 5) | kl;              // time index inside window
    float w = 0.5f * (1.0f - cosf(6.28318530717958647692f * (float)n / (float)NFFT));
    int ph = (n * f) & (NFFT - 1);        // exact integer phase reduction
    float th = (float)ph * (6.28318530717958647692f / (float)NFFT);
    float s, cc;
    sincosf(th, &s, &cc);
    Wr[idx] = (_Float16)(w * cc);
    Wi[idx] = (_Float16)(-w * s);
}

// ---------------- kernel 1: STFT as f16 WMMA GEMM + fused spec/ipd/pan ------
// Grid: (ceil(NFRAMES/64), ceil(FBINS/64)=9, BATCH). Block: 256 (8 waves).
// Block tile: 64 frames x 64 bins, both stereo channels.
// Waves: wm = wave&3 (frame 16-subtile), wn = wave>>2 (owns 2 bin 16-subtiles).
__global__ __launch_bounds__(256) void stft_wmma(const float* __restrict__ x,
                                                 const _Float16* __restrict__ Wr,
                                                 const _Float16* __restrict__ Wi,
                                                 float* __restrict__ out,
                                                 float* __restrict__ mag) {
    // 16KB LDS, aliased: GEMM staging in phase 1, f32 output tile in phase 2.
    __shared__ __align__(16) unsigned char smem[16384];
    _Float16* const ALs = (_Float16*)smem;        // 64x32 f16, left  (4KB)
    _Float16* const ARs = ALs + 2048;             // 64x32 f16, right (4KB)
    _Float16* const BRs = ARs + 2048;             // 64(n)x32(k) f16 cos (4KB)
    _Float16* const BIs = BRs + 2048;             // 64(n)x32(k) f16 sin (4KB)
    float*    const OT  = (float*)smem;           // 64(f)x64(t) f32 (16KB)

    const int tid  = threadIdx.x;
    const int b    = blockIdx.z;
    const int t0   = blockIdx.x * 64;
    const int f0   = blockIdx.y * 64;
    const float* xl = x + (size_t)b * 2 * T_IN;
    const float* xr = xl + T_IN;

    const int lane = tid & 31;
    const int wave = tid >> 5;
    const int wm   = wave & 3;
    const int wn   = wave >> 2;

    v8f accReL[2] = {}; v8f accImL[2] = {};
    v8f accReR[2] = {}; v8f accImR[2] = {};

    for (int kk = 0; kk < NFFT; kk += 32) {
        __syncthreads();   // previous iteration's fragment reads done before refill

        // ---- stage B: async copy (global -> LDS), one b128 per thread/plane.
        // Chunk-transposed layout makes this a straight contiguous copy.
        {
            const size_t choff = (size_t)(kk >> 5) * CHUNK_ELEMS + (size_t)f0 * 32;
            const _Float16* gR = Wr + choff + (size_t)tid * 8;   // 8 f16 = 16B
            const _Float16* gI = Wi + choff + (size_t)tid * 8;
            unsigned int ldsR = (unsigned int)(size_t)(BRs + tid * 8);
            unsigned int ldsI = (unsigned int)(size_t)(BIs + tid * 8);
            asm volatile("global_load_async_to_lds_b128 %0, %1, off"
                         :: "v"(ldsR), "v"((unsigned long long)(size_t)gR) : "memory");
            asm volatile("global_load_async_to_lds_b128 %0, %1, off"
                         :: "v"(ldsI), "v"((unsigned long long)(size_t)gI) : "memory");
        }

        // prefetch next twiddle chunk into cache (global_prefetch_b8)
        if (kk + 32 < NFFT)
            __builtin_prefetch(Wr + (size_t)((kk >> 5) + 1) * CHUNK_ELEMS +
                               (size_t)f0 * 32 + tid * 8, 0, 0);

        // ---- stage A: gather frames (reflect pad), f32->f16, packed b32 stores.
        for (int e = tid; e < 1024; e += 256) {       // 1024 pairs = 64x32 elems
            int m  = e >> 4;
            int jp = (e & 15) << 1;
            int t  = t0 + m;
            float vl0 = 0.f, vl1 = 0.f, vr0 = 0.f, vr1 = 0.f;
            if (t < NFRAMES) {
                int s0 = t * HOP + kk + jp - PADH;
                int s1 = s0 + 1;
                if (s0 < 0)          s0 = -s0;
                else if (s0 >= T_IN) s0 = 2 * T_IN - 2 - s0;
                if (s1 < 0)          s1 = -s1;
                else if (s1 >= T_IN) s1 = 2 * T_IN - 2 - s1;
                vl0 = xl[s0]; vl1 = xl[s1];
                vr0 = xr[s0]; vr1 = xr[s1];
            }
            union { _Float16 h[2]; unsigned int u; } pl, pr;
            pl.h[0] = (_Float16)vl0; pl.h[1] = (_Float16)vl1;
            pr.h[0] = (_Float16)vr0; pr.h[1] = (_Float16)vr1;
            *(unsigned int*)&ALs[m * 32 + jp] = pl.u;
            *(unsigned int*)&ARs[m * 32 + jp] = pr.u;
        }

        asm volatile("s_wait_asynccnt 0" ::: "memory");
        __syncthreads();

        // ---- fragments. A (16-bit 16x32, ISA 7.12.2): lane<16 K{0..7,16..23},
        // lane>=16 K{8..15,24..31}, M = lane%16.
        const int mrow = (wm * 16 + (lane & 15)) * 32;
        const int ka   = (lane < 16) ? 0 : 8;
        FragAB aL, aR;
        aL.f4[0] = *(const float4*)&ALs[mrow + ka];
        aL.f4[1] = *(const float4*)&ALs[mrow + ka + 16];
        aR.f4[0] = *(const float4*)&ARs[mrow + ka];
        aR.f4[1] = *(const float4*)&ARs[mrow + ka + 16];

        const int kb = (lane < 16) ? 0 : 8;    // B: lane<16 K{0..7,16..23}, else K{8..15,24..31}
#pragma unroll
        for (int s = 0; s < 2; ++s) {
            const int nrow = ((wn * 2 + s) * 16 + (lane & 15)) * 32;
            FragAB br_, bi_;
            br_.f4[0] = *(const float4*)&BRs[nrow + kb];
            br_.f4[1] = *(const float4*)&BRs[nrow + kb + 16];
            bi_.f4[0] = *(const float4*)&BIs[nrow + kb];
            bi_.f4[1] = *(const float4*)&BIs[nrow + kb + 16];

            accReL[s] = __builtin_amdgcn_wmma_f32_16x16x32_f16(false, aL.v, false, br_.v,
                                                               (short)0, accReL[s], false, false);
            accImL[s] = __builtin_amdgcn_wmma_f32_16x16x32_f16(false, aL.v, false, bi_.v,
                                                               (short)0, accImL[s], false, false);
            accReR[s] = __builtin_amdgcn_wmma_f32_16x16x32_f16(false, aR.v, false, br_.v,
                                                               (short)0, accReR[s], false, false);
            accImR[s] = __builtin_amdgcn_wmma_f32_16x16x32_f16(false, aR.v, false, bi_.v,
                                                               (short)0, accImR[s], false, false);
        }
    }

    // ---- epilogue: per output plane, scatter into a 64x64 f32 LDS tile then
    // write back t-contiguous float4s (full cacheline utilization).
    const size_t chs = (size_t)FBINS * NFRAMES;
    const int chn[4] = {0, 1, 3, 4};          // out channels; plane 4 -> mag
    const int flr  = tid >> 2;                 // readback row (f local)
    const int tseg = (tid & 3) << 4;           // readback t segment (16 floats)

#pragma unroll
    for (int plane = 0; plane < 5; ++plane) {
        __syncthreads();   // tile free (also covers GEMM->epilogue alias)
#pragma unroll
        for (int s = 0; s < 2; ++s) {
            const int fl = wn * 32 + s * 16 + (lane & 15);
#pragma unroll
            for (int r = 0; r < 8; ++r) {
                const int tl = wm * 16 + ((lane < 16) ? r : r + 8);
                float reL = accReL[s][r], imL = accImL[s][r];
                float reR = accReR[s][r], imR = accImR[s][r];
                float v;
                if (plane == 0)      v = 0.5f * (reL + reR);
                else if (plane == 1) v = 0.5f * (imL + imR);
                else if (plane == 2) v = atan2f(imL, reL) - atan2f(imR, reR);
                else if (plane == 3) {
                    float le = reL * reL + imL * imL;
                    float re2 = reR * reR + imR * imR;
                    v = (le - re2) / (le + re2 + 1e-10f);
                } else {
                    float re = 0.5f * (reL + reR);
                    float im = 0.5f * (imL + imR);
                    v = sqrtf(re * re + im * im);
                }
                OT[fl * 64 + tl] = v;
            }
        }
        __syncthreads();
        const int f = f0 + flr;
        if (f < FBINS) {
            float* dst = (plane < 4 ? out + (size_t)b * 6 * chs + (size_t)chn[plane] * chs
                                    : mag + (size_t)b * chs)
                         + (size_t)f * NFRAMES + t0 + tseg;
            const float* src = OT + flr * 64 + tseg;
            if (t0 + 64 <= NFRAMES) {
#pragma unroll
                for (int q = 0; q < 4; ++q)
                    *(float4*)(dst + q * 4) = *(const float4*)(src + q * 4);
            } else {
#pragma unroll
                for (int j = 0; j < 16; ++j)
                    if (t0 + tseg + j < NFRAMES) dst[j] = src[j];
            }
        }
    }
}

// ---------------- kernel 2: harmonic (HPS) + psychoacoustic planes ----------
// mag (62 MB) fits the 192 MB L2, so plain global reads with reuse are fine.
__global__ __launch_bounds__(256) void planes_kernel(const float* __restrict__ mag,
                                                     float* __restrict__ out) {
    const size_t chs = (size_t)FBINS * NFRAMES;
    size_t idx = (size_t)blockIdx.x * 256 + threadIdx.x;
    if (idx >= (size_t)BATCH * chs) return;
    int    b = (int)(idx / chs);
    size_t r = idx % chs;
    int    f = (int)(r / NFRAMES);
    int    t = (int)(r % NFRAMES);
    const float* mb = mag + (size_t)b * chs;

    // --- harmonic product spectrum via up(down(mag)) bilinear composition ---
    float hp = mb[(size_t)f * NFRAMES + t];
    const int FKs[4] = {FBINS / 2, FBINS / 3, FBINS / 4, FBINS / 5}; // 256,171,128,102
#pragma unroll
    for (int ki = 0; ki < 4; ++ki) {
        int Fk = FKs[ki];
        float jc = (f + 0.5f) * (float)Fk / (float)FBINS - 0.5f;
        int j0 = (int)floorf(jc);
        float wj = jc - (float)j0;
        float dv[2];
#pragma unroll
        for (int q = 0; q < 2; ++q) {
            int j = j0 + q;
            j = j < 0 ? 0 : (j >= Fk ? Fk - 1 : j);
            float ic = (j + 0.5f) * (float)FBINS / (float)Fk - 0.5f;
            int i0 = (int)floorf(ic);
            float wi = ic - (float)i0;
            int ia = i0     < 0 ? 0 : (i0     >= FBINS ? FBINS - 1 : i0);
            int ib = i0 + 1 < 0 ? 0 : (i0 + 1 >= FBINS ? FBINS - 1 : i0 + 1);
            dv[q] = (1.0f - wi) * mb[(size_t)ia * NFRAMES + t] +
                    wi          * mb[(size_t)ib * NFRAMES + t];
        }
        hp *= (1.0f - wj) * dv[0] + wj * dv[1];
    }

    // --- psychoacoustic: normalized Gaussian spread of 20*log10(mag+eps) ---
    float gw[21];
    float gsum = 0.0f;
#pragma unroll
    for (int j = 0; j < 21; ++j) {
        float d = (float)(j - 10);
        gw[j] = __expf(-0.5f * d * d);
        gsum += gw[j];
    }
    float acc = 0.0f;
#pragma unroll
    for (int j = -10; j <= 10; ++j) {
        int ff = f + j;
        if (ff >= 0 && ff < FBINS) {          // zero padding == skip tap
            float v = 20.0f * log10f(mb[(size_t)ff * NFRAMES + t] + 1e-10f);
            acc += gw[j + 10] * v;
        }
    }
    acc = acc / gsum - 20.0f;

    float* ob = out + (size_t)b * 6 * chs + r;
    ob[2 * chs] = hp;    // harmonic plane
    ob[5 * chs] = acc;   // psychoacoustic plane
}

// ---------------- launcher ----------------
extern "C" void kernel_launch(void* const* d_in, const int* in_sizes, int n_in,
                              void* d_out, int out_size, void* d_ws, size_t ws_size,
                              hipStream_t stream) {
    const float* x = (const float*)d_in[0];     // (16, 2, 480000) f32
    float* out = (float*)d_out;                 // (16, 6, 513, 1876) f32

    // workspace: f16 twiddles (2 * 1024*576, chunk-transposed) then f32 mag
    _Float16* Wr = (_Float16*)d_ws;
    _Float16* Wi = Wr + (size_t)NFFT * NPAD;
    float*    mg = (float*)(Wi + (size_t)NFFT * NPAD);

    init_twiddles<<<(NFFT * NPAD + 255) / 256, 256, 0, stream>>>(Wr, Wi);

    dim3 grid((NFRAMES + 63) / 64, (FBINS + 63) / 64, BATCH);   // (30, 9, 16)
    stft_wmma<<<grid, 256, 0, stream>>>(x, Wr, Wi, out, mg);

    size_t total = (size_t)BATCH * FBINS * NFRAMES;
    planes_kernel<<<(unsigned)((total + 255) / 256), 256, 0, stream>>>(mg, out);
}
